// MoEFFN_9234179687044
// MI455X (gfx1250) — matrix operand order repaired and checked by reading, not compiled, not run on hardware
//
#include <hip/hip_runtime.h>
#include <hip/hip_bf16.h>
#include <cstdint>

typedef __attribute__((ext_vector_type(16))) __bf16 bfrag;
typedef __attribute__((ext_vector_type(8)))  float  ffrag;
typedef __attribute__((ext_vector_type(4))) unsigned int u32x4;
typedef __attribute__((ext_vector_type(8))) int i32x8;
typedef __attribute__((ext_vector_type(4))) int i32x4;

constexpr int Bn = 4, Sn = 2048, Dn = 1024, Fn = 4096, En = 8, Kn = 2;
constexpr int NT  = Bn * Sn;      // 8192 tokens
constexpr int NE  = NT * Kn;      // 16384 routed entries
constexpr int CAP = NE;           // worst-case per-expert capacity
constexpr int TM = 128, TN = 128, TK = 32;
constexpr int LDA = TK + 8;       // 40 elems -> 80B rows (16B aligned)

// --- Tensor Data Mover availability (device pass only; this toolchain: 6-arg builtin) ---
#if defined(__HIP_DEVICE_COMPILE__) && defined(__has_builtin)
#if __has_builtin(__builtin_amdgcn_tensor_load_to_lds) && __has_builtin(__builtin_amdgcn_s_wait_tensorcnt)
#define USE_TDM 1
#endif
#endif
#ifndef USE_TDM
#define USE_TDM 0
#endif

// ---------------- fp32 -> bf16 conversion (no transpose; for hidden) ----------------
__global__ void cvt_kernel(const float* __restrict__ src, __bf16* __restrict__ dst, int n) {
    int i = (blockIdx.x * blockDim.x + threadIdx.x) * 4;
    if (i + 3 < n) {
        float4 v = *(const float4*)(src + i);
        union { __bf16 h[4]; unsigned long long u; } o;
        o.h[0] = (__bf16)v.x; o.h[1] = (__bf16)v.y;
        o.h[2] = (__bf16)v.z; o.h[3] = (__bf16)v.w;
        *(unsigned long long*)(dst + i) = o.u;
    }
}

// ------------- fp32 -> bf16 convert + transpose: src[R][C] -> dst[C][R] per expert -------------
__global__ void cvtT_kernel(const float* __restrict__ src, __bf16* __restrict__ dst,
                            int R, int C) {
    __shared__ float t[32][33];
    const float* s = src + (size_t)blockIdx.z * R * C;
    __bf16* d = dst + (size_t)blockIdx.z * R * C;
    int c0 = blockIdx.x * 32, r0 = blockIdx.y * 32;
    int tx = threadIdx.x & 31, ty = threadIdx.x >> 5;   // 8 rows per pass
#pragma unroll
    for (int i = 0; i < 32; i += 8)
        t[ty + i][tx] = s[(size_t)(r0 + ty + i) * C + c0 + tx];
    __syncthreads();
#pragma unroll
    for (int i = 0; i < 32; i += 8)
        d[(size_t)(c0 + ty + i) * R + r0 + tx] = (__bf16)t[tx][ty + i];
}

// ---------------- routing ----------------
__global__ void route_init(int* cnt) {
    if (threadIdx.x < En) cnt[threadIdx.x] = 0;
}

__global__ void route_kernel(const int* __restrict__ idx, const float* __restrict__ wts,
                             int* __restrict__ cnt, int* __restrict__ slots,
                             float* __restrict__ gates) {
    int t = blockIdx.x * blockDim.x + threadIdx.x;
    if (t < NT) {
#pragma unroll
        for (int k = 0; k < Kn; ++k) {
            int e = idx[t * Kn + k] & (En - 1);
            float wv = wts[t * Kn + k];
            int p = atomicAdd(&cnt[e], 1);
            slots[e * CAP + p] = t * Kn + k;   // entry id encodes (token, slot)
            gates[e * CAP + p] = wv;
        }
    }
}

// ---------------- helpers ----------------
__device__ __forceinline__ float gelu_tanh(float x) {
    float x3 = x * x * x;
    return 0.5f * x * (1.0f + tanhf(0.7978845608028654f * (x + 0.044715f * x3)));
}

#if USE_TDM
// Build D# for a 2D bf16 tile: TK (contiguous) x TN rows, row stride = stride0 elems,
// padded in LDS by 16B after every 64B row (-> 80B = LDA*2 row pitch).
__device__ __forceinline__ void tdm_load_tile(unsigned lds_off, const __bf16* g,
                                              unsigned d0, unsigned d1, unsigned stride0) {
    unsigned long long ga = (unsigned long long)(uintptr_t)g;
    u32x4 g0;
    g0[0] = 1u;                                   // count=1 (valid), user mode
    g0[1] = lds_off;                              // LDS byte address
    g0[2] = (unsigned)ga;                         // global_addr[31:0]
    g0[3] = (unsigned)((ga >> 32) & 0x01FFFFFFull) | (2u << 30);  // addr[56:32] | type=2
    i32x8 g1;
    g1[0] = (int)((1u << 16)     // data_size = 2 bytes
                | (1u << 20)     // pad_enable
                | (3u << 22)     // pad_interval: 16 DWORDs (one 64B tile row)
                | (3u << 25));   // pad_amount: 4 DWORDs (16B)
    g1[1] = (int)((d0 & 0xFFFFu) << 16);                          // tensor_dim0[15:0]
    g1[2] = (int)((d0 >> 16) | ((d1 & 0xFFFFu) << 16));           // dim0 hi | dim1 lo
    g1[3] = (int)((d1 >> 16) | ((unsigned)TK << 16));             // dim1 hi | tile_dim0=32
    g1[4] = (int)(unsigned)TN;                                    // tile_dim1=128, tile_dim2=0
    g1[5] = (int)stride0;                                         // tensor_dim0_stride lo
    g1[6] = 0;                                                    // stride0 hi | stride1 lo
    g1[7] = 0;
    i32x4 z4 = {0, 0, 0, 0};
    i32x8 z8 = {0, 0, 0, 0, 0, 0, 0, 0};
    // 6-arg form on this toolchain: (g0, g1, g2, g3, extra, cpol)
    __builtin_amdgcn_tensor_load_to_lds(g0, g1, z4, z4, z8, 0);
}
#endif

// ---------------- grouped GEMM (MODE 0: X@W1t -> gelu -> H bf16; MODE 1: H@W2t -> *gate -> Y f32) ----------------
template <int MODE>
__global__ void __launch_bounds__(256)
moe_gemm(const __bf16* __restrict__ Abase, const __bf16* __restrict__ Wt,
         const float* __restrict__ bias, const int* __restrict__ cnt,
         const int* __restrict__ slots, const float* __restrict__ gates,
         __bf16* __restrict__ Hout, float* __restrict__ Yout) {
    constexpr int KD = (MODE == 0) ? Dn : Fn;   // reduction dim
    constexpr int ND = (MODE == 0) ? Fn : Dn;   // output dim

    const int e = blockIdx.z, mt = blockIdx.x, nt = blockIdx.y;
    const int c = cnt[e];
    const int row0 = mt * TM;
    if (row0 >= c) return;
    const int valid = min(TM, c - row0);

    __shared__ __bf16 As[2][TM * LDA];
    __shared__ __bf16 Bs[2][TN * LDA];
    __shared__ int slotS[TM];
    __shared__ float gateS[TM];

    const int tid = threadIdx.x;
    if (tid < TM) {
        int j = min(row0 + tid, c - 1);
        slotS[tid] = slots[e * CAP + j];
        gateS[tid] = (MODE == 1) ? gates[e * CAP + j] : 0.0f;
    }
    __syncthreads();

    const int w = tid >> 5, lane = tid & 31;
    const int wm = w >> 2, wn = w & 3;           // 2 x 4 wave grid, wave tile 64x32
    const int lm = lane & 15, lh = lane >> 4;

    const int r = tid >> 1, half = tid & 1;      // staging: thread -> (row, 32B half)
    const int arow = (MODE == 0) ? (slotS[r] >> 1) : slotS[r];
    const __bf16* aSrc = Abase + (size_t)arow * KD + half * 16;
    const __bf16* bTile = Wt + ((size_t)e * ND + (size_t)nt * TN) * KD;  // tile origin
#if !USE_TDM
    const __bf16* bSrc = bTile + (size_t)r * KD + half * 16;
#endif

    ffrag acc[4][2] = {};

    // ---- prologue: stage k-block 0 into buffer 0 ----
#if USE_TDM
    if (w == 0)
        tdm_load_tile((unsigned)(uintptr_t)&Bs[0][0], bTile, (unsigned)KD, (unsigned)ND, (unsigned)KD);
#else
    {
        uint4 b0 = *(const uint4*)(bSrc);
        uint4 b1 = *(const uint4*)(bSrc + 8);
        *(uint4*)&Bs[0][r * LDA + half * 16]     = b0;
        *(uint4*)&Bs[0][r * LDA + half * 16 + 8] = b1;
    }
#endif
    {
        uint4 a0 = *(const uint4*)(aSrc);
        uint4 a1 = *(const uint4*)(aSrc + 8);
        *(uint4*)&As[0][r * LDA + half * 16]     = a0;
        *(uint4*)&As[0][r * LDA + half * 16 + 8] = a1;
    }

    int buf = 0;
    for (int kk = 0; kk < KD; kk += TK) {
#if USE_TDM
        if (w == 0) __builtin_amdgcn_s_wait_tensorcnt(0);
#endif
        __syncthreads();
        const bool more = (kk + TK) < KD;

        uint4 a0, a1;
#if !USE_TDM
        uint4 b0, b1;
#endif
        if (more) {
#if USE_TDM
            if (w == 0)
                tdm_load_tile((unsigned)(uintptr_t)&Bs[buf ^ 1][0], bTile + kk + TK,
                              (unsigned)KD, (unsigned)ND, (unsigned)KD);
#else
            b0 = *(const uint4*)(bSrc + kk + TK);
            b1 = *(const uint4*)(bSrc + kk + TK + 8);
#endif
            a0 = *(const uint4*)(aSrc + kk + TK);
            a1 = *(const uint4*)(aSrc + kk + TK + 8);
        }

        // ---- WMMA on current buffer: all fragments are two aligned ds_load_b128 ----
        bfrag af[4], bf[2];
#pragma unroll
        for (int mf = 0; mf < 4; ++mf) {
            // A elems 0..7 -> K = 8*lh + j ; elems 8..15 -> K = 16 + 8*lh + j
            const __bf16* p = &As[buf][(wm * 64 + mf * 16 + lm) * LDA];
            ((uint4*)&af[mf])[0] = *(const uint4*)(p + lh * 8);
            ((uint4*)&af[mf])[1] = *(const uint4*)(p + 16 + lh * 8);
        }
#pragma unroll
        for (int nf = 0; nf < 2; ++nf) {
            // B elems 0..7 -> K = 8*lh + j ; elems 8..15 -> K = 16 + 8*lh + j
            const __bf16* p = &Bs[buf][(wn * 32 + nf * 16 + lm) * LDA];
            ((uint4*)&bf[nf])[0] = *(const uint4*)(p + lh * 8);
            ((uint4*)&bf[nf])[1] = *(const uint4*)(p + 16 + lh * 8);
        }
#pragma unroll
        for (int mf = 0; mf < 4; ++mf)
#pragma unroll
            for (int nf = 0; nf < 2; ++nf)
                acc[mf][nf] = __builtin_amdgcn_wmma_f32_16x16x32_bf16(
                    false, af[mf], false, bf[nf], (short)0, acc[mf][nf], false, false);

        if (more) {
            *(uint4*)&As[buf ^ 1][r * LDA + half * 16]     = a0;
            *(uint4*)&As[buf ^ 1][r * LDA + half * 16 + 8] = a1;
#if !USE_TDM
            *(uint4*)&Bs[buf ^ 1][r * LDA + half * 16]     = b0;
            *(uint4*)&Bs[buf ^ 1][r * LDA + half * 16 + 8] = b1;
#endif
        }
        buf ^= 1;
    }

    // ---- epilogue ----
#pragma unroll
    for (int mf = 0; mf < 4; ++mf)
#pragma unroll
        for (int nf = 0; nf < 2; ++nf) {
            int ncol = nt * TN + wn * 32 + nf * 16 + lm;
            float bv = bias[(size_t)e * ND + ncol];
#pragma unroll
            for (int r8 = 0; r8 < 8; ++r8) {
                int rr = wm * 64 + mf * 16 + r8 + 8 * lh;  // C: lanes 16-31 hold M+8
                if (rr < valid) {
                    float v = acc[mf][nf][r8] + bv;
                    if (MODE == 0) {
                        Hout[(size_t)slotS[rr] * Fn + ncol] = (__bf16)gelu_tanh(v);
                    } else {
                        Yout[(size_t)slotS[rr] * Dn + ncol] = v * gateS[rr];
                    }
                }
            }
        }
}

// ---------------- combine: out[t] = Y[2t] + Y[2t+1] (deterministic) ----------------
__global__ void combine_kernel(const float* __restrict__ Y, float* __restrict__ out) {
    size_t i = ((size_t)blockIdx.x * blockDim.x + threadIdx.x) * 4;
    if (i < (size_t)NT * Dn) {
        size_t t = i / Dn, d = i % Dn;
        float4 y0 = *(const float4*)(Y + (2 * t) * (size_t)Dn + d);
        float4 y1 = *(const float4*)(Y + (2 * t + 1) * (size_t)Dn + d);
        float4 o;
        o.x = y0.x + y1.x; o.y = y0.y + y1.y;
        o.z = y0.z + y1.z; o.w = y0.w + y1.w;
        *(float4*)(out + i) = o;
    }
}

// ---------------- launch ----------------
extern "C" void kernel_launch(void* const* d_in, const int* in_sizes, int n_in,
                              void* d_out, int out_size, void* d_ws, size_t ws_size,
                              hipStream_t stream) {
    (void)in_sizes; (void)n_in; (void)out_size; (void)ws_size;
    const float* hidden = (const float*)d_in[0];
    const int*   tki    = (const int*)d_in[1];
    const float* tkw    = (const float*)d_in[2];
    const float* w1     = (const float*)d_in[3];
    const float* b1     = (const float*)d_in[4];
    const float* w2     = (const float*)d_in[5];
    const float* b2     = (const float*)d_in[6];
    float* out = (float*)d_out;

    char* ws = (char*)d_ws;
    size_t off = 0;
    auto alloc = [&](size_t bytes) -> void* {
        void* p = ws + off;
        off = (off + bytes + 255) & ~(size_t)255;
        return p;
    };
    __bf16* W1t  = (__bf16*)alloc((size_t)En * Dn * Fn * 2);  // [E][F][D] bf16
    __bf16* W2t  = (__bf16*)alloc((size_t)En * Fn * Dn * 2);  // [E][D][F] bf16
    __bf16* Xb   = (__bf16*)alloc((size_t)NT * Dn * 2);
    __bf16* H    = (__bf16*)alloc((size_t)NE * Fn * 2);
    float*  Y    = (float*)alloc((size_t)NE * Dn * 4);
    int*    cnt  = (int*)alloc(En * 4);
    int*    slot = (int*)alloc((size_t)En * CAP * 4);
    float*  gate = (float*)alloc((size_t)En * CAP * 4);

    // weights: convert + transpose to K-major bf16
    dim3 t1(Fn / 32, Dn / 32, En);
    cvtT_kernel<<<t1, 256, 0, stream>>>(w1, W1t, Dn, Fn);
    dim3 t2(Dn / 32, Fn / 32, En);
    cvtT_kernel<<<t2, 256, 0, stream>>>(w2, W2t, Fn, Dn);
    const int nx = NT * Dn;
    cvt_kernel<<<nx / 1024, 256, 0, stream>>>(hidden, Xb, nx);

    route_init<<<1, 32, 0, stream>>>(cnt);
    route_kernel<<<NT / 256, 256, 0, stream>>>(tki, tkw, cnt, slot, gate);

    dim3 g1(CAP / TM, Fn / TN, En);     // (128, 32, 8); inactive tiles exit early
    moe_gemm<0><<<g1, 256, 0, stream>>>(Xb, W1t, b1, cnt, slot, nullptr, H, nullptr);

    dim3 g2(CAP / TM, Dn / TN, En);     // (128, 8, 8)
    moe_gemm<1><<<g2, 256, 0, stream>>>(H, W2t, b2, cnt, slot, gate, nullptr, Y);

    combine_kernel<<<(NT * Dn / 4) / 256, 256, 0, stream>>>(Y, out);
}
